// ParallelMultiHeadedAttention_25305947308728
// MI455X (gfx1250) — matrix operand order repaired and checked by reading, not compiled, not run on hardware
//
#include <hip/hip_runtime.h>
#include <stddef.h>

// ---------------------------------------------------------------------------
// Problem constants (match reference)
// ---------------------------------------------------------------------------
constexpr int Bc   = 2;
constexpr int Sc   = 2048;
constexpr int HIDc = 2048;
constexpr int Hc   = 16;
constexpr int Dc   = 128;
constexpr int KVc  = 4;
constexpr int REPc = Hc / KVc;

typedef __attribute__((ext_vector_type(16))) __bf16       v16bf;
typedef __attribute__((ext_vector_type(8)))  float        v8f;
typedef __attribute__((ext_vector_type(4)))  unsigned int u32x4;
typedef __attribute__((ext_vector_type(8)))  int          i32x8;

#define DEVFN __device__ __forceinline__

DEVFN unsigned short f2bf(float f) {
  union { float f; unsigned u; } c; c.f = f;
  unsigned u = c.u + 0x7FFFu + ((c.u >> 16) & 1u);  // round-to-nearest-even
  return (unsigned short)(u >> 16);
}
DEVFN float bf2f(unsigned short h) {
  union { unsigned u; float f; } c; c.u = ((unsigned)h) << 16;
  return c.f;
}

union FragBF {              // one WMMA 16x16x32 bf16 operand (8 VGPRs)
  v16bf v;
  u32x4 u[2];
};

DEVFN v8f wmma_bf16(const v16bf a, const v16bf b, const v8f c) {
  return __builtin_amdgcn_wmma_f32_16x16x32_bf16(
      false, a, false, b, (short)0, c, false, false);
}

// ---------------------------------------------------------------------------
// Generic tiled GEMM with double-buffered LDS:  C[M,N] = A[M,K] @ W[K,N] + bias
//   Block 256 thr = 8 waves; tile 128x128x32; wave tile 32x64 (2x4 WMMA frags).
//   One barrier per K-step; next tile staged while current tile is multiplied.
// ---------------------------------------------------------------------------
constexpr int GBM = 128, GBN = 128, GBK = 32;
constexpr int GLA = GBK + 8;   // 40 shorts (80B, 16B-aligned rows)
constexpr int GLB = GBK + 8;   // Bs stored N-major: Bs[n][k]

template <bool A_IS_BF16, bool OUT_BF16>
__global__ __launch_bounds__(256) void gemm_wmma(
    const void* __restrict__ Aptr, const float* __restrict__ W,
    const float* __restrict__ bias, void* __restrict__ Cptr,
    int M, int N, int K) {
  __shared__ __align__(16) unsigned short As[2][GBM * GLA];
  __shared__ __align__(16) unsigned short Bs[2][GBN * GLB];

  const int tid  = threadIdx.x;
  const int lane = tid & 31;
  const int wave = tid >> 5;           // 0..7
  const int wm   = wave & 3;           // 4 waves along M (x32 rows)
  const int wn   = wave >> 2;          // 2 waves along N (x64 cols)
  const int lm   = lane & 15;
  const int hh   = lane >> 4;          // lane half
  const int bm   = blockIdx.y * GBM;
  const int bn   = blockIdx.x * GBN;

  auto stage = [&](int k0, int buf) {
    {  // A tile (128x32) -> bf16 row-major
      const int r  = tid >> 1;
      const int cb = (tid & 1) * 16;
      if constexpr (A_IS_BF16) {
        const unsigned short* a = (const unsigned short*)Aptr;
        const size_t off = (size_t)(bm + r) * K + k0 + cb;
        *(u32x4*)&As[buf][r * GLA + cb]     = *(const u32x4*)&a[off];
        *(u32x4*)&As[buf][r * GLA + cb + 8] = *(const u32x4*)&a[off + 8];
      } else {
        const float* a = (const float*)Aptr;
        const size_t off = (size_t)(bm + r) * K + k0 + cb;
#pragma unroll
        for (int j = 0; j < 16; ++j)
          As[buf][r * GLA + cb + j] = f2bf(a[off + j]);
      }
    }
    {  // B tile (32x128) f32 -> bf16, transposed to Bs[n][k]
      const int kr = tid >> 3;
      const int cb = (tid & 7) * 16;
      const float* wrow = &W[(size_t)(k0 + kr) * N + bn + cb];
#pragma unroll
      for (int j = 0; j < 16; ++j)
        Bs[buf][(cb + j) * GLB + kr] = f2bf(wrow[j]);
    }
  };

  v8f acc[2][4] = {};
  stage(0, 0);
  int cur = 0;

  for (int k0 = 0; k0 < K; k0 += GBK) {
    __syncthreads();
    if (k0 + GBK < K) stage(k0 + GBK, cur ^ 1);  // overlap with WMMA below

    FragBF fa[2], fb[4];
#pragma unroll
    for (int mi = 0; mi < 2; ++mi) {
      const int m = wm * 32 + mi * 16 + lm;
      fa[mi].u[0] = *(const u32x4*)&As[cur][m * GLA + hh * 8];
      fa[mi].u[1] = *(const u32x4*)&As[cur][m * GLA + 16 + hh * 8];
    }
#pragma unroll
    for (int ni = 0; ni < 4; ++ni) {
      const int n = wn * 64 + ni * 16 + lm;
      fb[ni].u[0] = *(const u32x4*)&Bs[cur][n * GLB + hh * 8];
      fb[ni].u[1] = *(const u32x4*)&Bs[cur][n * GLB + 16 + hh * 8];
    }
#pragma unroll
    for (int mi = 0; mi < 2; ++mi)
#pragma unroll
      for (int ni = 0; ni < 4; ++ni)
        acc[mi][ni] = wmma_bf16(fa[mi].v, fb[ni].v, acc[mi][ni]);
    cur ^= 1;
  }

#pragma unroll
  for (int mi = 0; mi < 2; ++mi) {
#pragma unroll
    for (int ni = 0; ni < 4; ++ni) {
      const int col = bn + wn * 64 + ni * 16 + lm;
      const float bv = bias ? bias[col] : 0.0f;
#pragma unroll
      for (int i = 0; i < 8; ++i) {
        const int row = bm + wm * 32 + mi * 16 + i + 8 * hh;
        const float val = acc[mi][ni][i] + bv;
        if constexpr (OUT_BF16)
          ((unsigned short*)Cptr)[(size_t)row * N + col] = f2bf(val);
        else
          ((float*)Cptr)[(size_t)row * N + col] = val;
      }
    }
  }
}

// ---------------------------------------------------------------------------
// RoPE on bf16 buffer laid out (B, S, nh, D); interleaved even/odd pairs.
// ---------------------------------------------------------------------------
__global__ __launch_bounds__(256) void rope_kernel(
    unsigned short* __restrict__ x, const float* __restrict__ cosf,
    const float* __restrict__ sinf, int nh) {
  const size_t idx = (size_t)blockIdx.x * blockDim.x + threadIdx.x;
  const size_t total = (size_t)Bc * Sc * nh * (Dc / 2);
  if (idx >= total) return;
  const int d2 = (int)(idx & 63);
  size_t t = idx >> 6;
  const int hh = (int)(t % nh); t /= nh;
  const int s  = (int)(t % Sc);
  const int b  = (int)(t / Sc);
  const size_t off = (((size_t)b * Sc + s) * nh + hh) * Dc + 2 * d2;
  const float e = bf2f(x[off]), o = bf2f(x[off + 1]);
  const float c = cosf[s * 64 + d2], sn = sinf[s * 64 + d2];
  x[off]     = f2bf(e * c - o * sn);
  x[off + 1] = f2bf(e * sn + o * c);
}

// ---------------------------------------------------------------------------
// Flash attention (causal, GQA). Grid (S/64, H, B); block 128 thr = 4 waves.
// K tile staged by the Tensor Data Mover (wave 0 issues tensor_load_to_lds
// with a padded-destination D#; TENSORcnt waited before the barrier).
// ---------------------------------------------------------------------------
constexpr int KLD = Dc + 8;    // 136 shorts: matches TDM pad (64DW rows + 4DW pad)
constexpr int VLD = 32 + 8;    // 40 shorts:  V^T tile row stride

__global__ __launch_bounds__(128) void flash_attn(
    const unsigned short* __restrict__ qb, const unsigned short* __restrict__ kb,
    const unsigned short* __restrict__ vb, unsigned short* __restrict__ ctx) {
  __shared__ __align__(16) unsigned short Ks[32 * KLD];      // K: [key][d] (TDM dest)
  __shared__ __align__(16) unsigned short Vt[Dc * VLD];      // V^T: [d][key]
  __shared__ __align__(16) unsigned short Ps[4 * 16 * 32];   // per-wave P patch

  const int tid  = threadIdx.x;
  const int lane = tid & 31;
  const int wave = tid >> 5;     // 0..3
  const int lm   = lane & 15;
  const int hh   = lane >> 4;

  const int qblk = blockIdx.x;
  const int h    = blockIdx.y;
  const int b    = blockIdx.z;
  const int hk   = h / REPc;

  const unsigned short* Q  = qb + (size_t)b * Sc * Hc * Dc + (size_t)h * Dc;
  const unsigned short* Kg = kb + (size_t)b * Sc * KVc * Dc + (size_t)hk * Dc;
  const unsigned short* Vg = vb + (size_t)b * Sc * KVc * Dc + (size_t)hk * Dc;

  const int qrow0 = qblk * 64 + wave * 16;

  // Q held in registers: 4 A-fragments spanning D=128
  FragBF qf[4];
  {
    const size_t rowoff = (size_t)(qrow0 + lm) * (Hc * Dc);
#pragma unroll
    for (int kd = 0; kd < 4; ++kd) {
      qf[kd].u[0] = *(const u32x4*)&Q[rowoff + kd * 32 + hh * 8];
      qf[kd].u[1] = *(const u32x4*)&Q[rowoff + kd * 32 + 16 + hh * 8];
    }
  }

  v8f  U[8] = {};
  float mrow[8], lrow[8];
#pragma unroll
  for (int i = 0; i < 8; ++i) { mrow[i] = -1e30f; lrow[i] = 0.0f; }

  const float scale = 0.08838834764831845f;  // 1/sqrt(128)
  const int kend = qblk * 64 + 64;

  for (int kb0 = 0; kb0 < kend; kb0 += 32) {
    __syncthreads();

    // ---- TDM: stage K tile 32x128 bf16 -> Ks with 16B row padding ----
    if (wave == 0) {
      const unsigned long long ga =
          (unsigned long long)(const void*)(Kg + (size_t)kb0 * (KVc * Dc));
      u32x4 g0;
      g0[0] = 1u;                                     // count=1, user D#
      g0[1] = (unsigned)(unsigned long long)(&Ks[0]); // lds_addr (LDS offset)
      g0[2] = (unsigned)ga;                           // global_addr[31:0]
      g0[3] = (unsigned)((ga >> 32) & 0x01FFFFFFu) | (2u << 30);  // addr[56:32], type=2
      i32x8 g1;
      g1[0] = (int)((1u << 16)      // data_size = 2 bytes
                  | (1u << 20)      // pad_enable
                  | (5u << 22)      // pad_interval: every 64 DWORDs (one 128-elem row)
                  | (3u << 25));    // pad_amount: 4 DWORDs (8 shorts) -> stride 136
      g1[1] = (int)(128u << 16);    // tensor_dim0 = 128 (head slice width)
      g1[2] = (int)((unsigned)Sc << 16);  // tensor_dim1 = 2048
      g1[3] = (int)(128u << 16);    // tile_dim0 = 128
      g1[4] = 32;                   // tile_dim1 = 32
      g1[5] = KVc * Dc;             // tensor_dim0_stride = 512 elements
      g1[6] = 0;
      g1[7] = 0;
      asm volatile("tensor_load_to_lds %0, %1" :: "s"(g0), "s"(g1) : "memory");
    }

    // ---- stage V tile transposed: Vt[d][key] (all threads, overlaps TDM) ----
    {
      const int kr = tid >> 2;          // 0..31
      const int db = (tid & 3) * 32;    // 0,32,64,96
      const unsigned short* vs = &Vg[(size_t)(kb0 + kr) * (KVc * Dc) + db];
#pragma unroll
      for (int j = 0; j < 32; ++j)
        Vt[(db + j) * VLD + kr] = vs[j];
    }

    if (wave == 0) __builtin_amdgcn_s_wait_tensorcnt(0);
    __syncthreads();

    // Blocks entirely above the diagonal for this wave contribute nothing.
    const bool active = (kb0 <= qrow0 + 15);

    v8f sf[2] = {};
    if (active) {
      // ---- S = (Q K^T) * scale ----
#pragma unroll
      for (int ni = 0; ni < 2; ++ni) {
        const int n = ni * 16 + lm;
        FragBF kf;
#pragma unroll
        for (int kd = 0; kd < 4; ++kd) {
          kf.u[0] = *(const u32x4*)&Ks[n * KLD + kd * 32 + hh * 8];
          kf.u[1] = *(const u32x4*)&Ks[n * KLD + kd * 32 + 16 + hh * 8];
          sf[ni] = wmma_bf16(qf[kd].v, kf.v, sf[ni]);
        }
#pragma unroll
        for (int i = 0; i < 8; ++i) sf[ni][i] *= scale;
      }
      // ---- causal mask: only the diagonal block needs it ----
      if (kb0 + 31 > qrow0) {
#pragma unroll
        for (int ni = 0; ni < 2; ++ni) {
          const int colg = kb0 + ni * 16 + lm;
#pragma unroll
          for (int i = 0; i < 8; ++i) {
            const int rowg = qrow0 + i + 8 * hh;
            if (colg > rowg) sf[ni][i] += -10000.0f;
          }
        }
      }
      // ---- online softmax (row reductions across 16-lane halves) ----
#pragma unroll
      for (int i = 0; i < 8; ++i) {
        float rm = fmaxf(sf[0][i], sf[1][i]);
        rm = fmaxf(rm, __shfl_xor(rm, 1, 32));
        rm = fmaxf(rm, __shfl_xor(rm, 2, 32));
        rm = fmaxf(rm, __shfl_xor(rm, 4, 32));
        rm = fmaxf(rm, __shfl_xor(rm, 8, 32));
        const float mn  = fmaxf(mrow[i], rm);
        const float cor = __expf(mrow[i] - mn);
        const float p0  = __expf(sf[0][i] - mn);
        const float p1  = __expf(sf[1][i] - mn);
        float rs = p0 + p1;
        rs += __shfl_xor(rs, 1, 32);
        rs += __shfl_xor(rs, 2, 32);
        rs += __shfl_xor(rs, 4, 32);
        rs += __shfl_xor(rs, 8, 32);
        lrow[i] = lrow[i] * cor + rs;
        mrow[i] = mn;
        sf[0][i] = p0;
        sf[1][i] = p1;
#pragma unroll
        for (int nd = 0; nd < 8; ++nd) U[nd][i] *= cor;
      }
      // ---- C-layout -> A-layout for P via per-wave LDS patch ----
      unsigned short* Pw = &Ps[wave * 16 * 32];
#pragma unroll
      for (int ni = 0; ni < 2; ++ni)
#pragma unroll
        for (int i = 0; i < 8; ++i)
          Pw[(i + 8 * hh) * 32 + ni * 16 + lm] = f2bf(sf[ni][i]);
    }
    __syncthreads();

    if (active) {
      const unsigned short* Pw = &Ps[wave * 16 * 32];
      FragBF pf;
      pf.u[0] = *(const u32x4*)&Pw[lm * 32 + hh * 8];
      pf.u[1] = *(const u32x4*)&Pw[lm * 32 + 16 + hh * 8];
#pragma unroll
      for (int nd = 0; nd < 8; ++nd) {
        FragBF vf;
        const int n = nd * 16 + lm;
        vf.u[0] = *(const u32x4*)&Vt[n * VLD + hh * 8];
        vf.u[1] = *(const u32x4*)&Vt[n * VLD + 16 + hh * 8];
        U[nd] = wmma_bf16(pf.v, vf.v, U[nd]);
      }
    }
  }

  // ---- finalize: ctx[b, row, h, :] = U / l (bf16 for out-proj WMMA) ----
#pragma unroll
  for (int nd = 0; nd < 8; ++nd) {
#pragma unroll
    for (int i = 0; i < 8; ++i) {
      const int row = qrow0 + i + 8 * hh;
      const int col = nd * 16 + lm;
      const float o = U[nd][i] / lrow[i];
      ctx[((size_t)b * Sc + row) * (Hc * Dc) + (size_t)h * Dc + col] = f2bf(o);
    }
  }
}

// ---------------------------------------------------------------------------
// Launch
// ---------------------------------------------------------------------------
extern "C" void kernel_launch(void* const* d_in, const int* in_sizes, int n_in,
                              void* d_out, int out_size, void* d_ws,
                              size_t ws_size, hipStream_t stream) {
  const float* query = (const float*)d_in[0];
  const float* key   = (const float*)d_in[1];
  const float* value = (const float*)d_in[2];
  // d_in[3] = mask (causal * -10000) — applied analytically in-kernel
  const float* fcos  = (const float*)d_in[4];
  const float* fsin  = (const float*)d_in[5];
  const float* Wq = (const float*)d_in[6];
  const float* bq = (const float*)d_in[7];
  const float* Wk = (const float*)d_in[8];
  const float* bk = (const float*)d_in[9];
  const float* Wv = (const float*)d_in[10];
  const float* bv = (const float*)d_in[11];
  const float* Wo = (const float*)d_in[12];
  const float* bo = (const float*)d_in[13];

  unsigned short* qbuf = (unsigned short*)d_ws;
  unsigned short* kbuf = qbuf + (size_t)Bc * Sc * Hc * Dc;
  unsigned short* vbuf = kbuf + (size_t)Bc * Sc * KVc * Dc;
  unsigned short* ctx  = vbuf + (size_t)Bc * Sc * KVc * Dc;

  const int M = Bc * Sc;  // 4096

  gemm_wmma<false, true><<<dim3((Hc * Dc) / GBN, M / GBM), 256, 0, stream>>>(
      query, Wq, bq, qbuf, M, Hc * Dc, HIDc);
  gemm_wmma<false, true><<<dim3((KVc * Dc) / GBN, M / GBM), 256, 0, stream>>>(
      key, Wk, bk, kbuf, M, KVc * Dc, HIDc);
  gemm_wmma<false, true><<<dim3((KVc * Dc) / GBN, M / GBM), 256, 0, stream>>>(
      value, Wv, bv, vbuf, M, KVc * Dc, HIDc);

  {
    const size_t nq = (size_t)Bc * Sc * Hc * (Dc / 2);
    const size_t nk = (size_t)Bc * Sc * KVc * (Dc / 2);
    rope_kernel<<<dim3((unsigned)((nq + 255) / 256)), 256, 0, stream>>>(qbuf, fcos, fsin, Hc);
    rope_kernel<<<dim3((unsigned)((nk + 255) / 256)), 256, 0, stream>>>(kbuf, fcos, fsin, KVc);
  }

  flash_attn<<<dim3(Sc / 64, Hc, Bc), 128, 0, stream>>>(qbuf, kbuf, vbuf, ctx);

  gemm_wmma<true, false><<<dim3(HIDc / GBN, M / GBM), 256, 0, stream>>>(
      ctx, Wo, bo, (float*)d_out, M, HIDc, Hc * Dc);
}
